// HeadAttention_26963804684685
// MI455X (gfx1250) — matrix operand order repaired and checked by reading, not compiled, not run on hardware
//
#include <hip/hip_runtime.h>
#include <hip/hip_bf16.h>

typedef __attribute__((ext_vector_type(16))) __bf16 v16bf;
typedef __attribute__((ext_vector_type(8)))  __bf16 v8bf;
typedef __attribute__((ext_vector_type(8)))  float  v8f;
typedef __attribute__((ext_vector_type(4)))  float  f32x4;
typedef __attribute__((ext_vector_type(4)))  int    v4i;

#define L2E 1.4426950408889634f

#if defined(__gfx1250__) && __has_builtin(__builtin_amdgcn_global_load_async_to_lds_b128) && __has_builtin(__builtin_amdgcn_s_wait_asynccnt)
#define ATTN_ASYNC 1
#else
#define ATTN_ASYNC 0
#endif

__device__ __forceinline__ v8f wmma_bf16(v16bf a, v16bf b, v8f c) {
    // D = A(16x32 bf16) * B(32x16 bf16) + C(16x16 f32)
    return __builtin_amdgcn_wmma_f32_16x16x32_bf16(
        false, a, false, b, (short)0, c, false, false);
}

// 16-byte global -> LDS copy; async on CDNA5 (ASYNCcnt-tracked), sync fallback.
__device__ __forceinline__ void stage16(const __bf16* g, __bf16* l) {
#if ATTN_ASYNC
    __builtin_amdgcn_global_load_async_to_lds_b128(
        (__attribute__((address_space(1))) v4i*)g,
        (__attribute__((address_space(3))) v4i*)l, 0, 0);
#else
    *(f32x4*)l = *(const f32x4*)g;
#endif
}

// ---------------------------------------------------------------------------
// Kernel 0: pre-swizzle W (f32 [1024][64]) into bf16 hi/lo WMMA B-fragments.
// B_frag[j] = W[kchunk*32 + (lane>>4)*8 + j + (j>=8 ? 8 : 0)][ntile*16 + (lane&15)]
// ---------------------------------------------------------------------------
__global__ void __launch_bounds__(256) prep_w(
    const float* __restrict__ Wq, const float* __restrict__ Wk,
    const float* __restrict__ Wv, __bf16* __restrict__ whi,
    __bf16* __restrict__ wlo) {
    int tid = blockIdx.x * blockDim.x + threadIdx.x;   // 12288 total
    if (tid >= 3 * 4 * 32 * 32) return;
    int lane = tid & 31;
    int c    = (tid >> 5) & 31;
    int t    = (tid >> 10) & 3;
    int m    = tid >> 12;                              // 0..2
    const float* W = (m == 0) ? Wq : (m == 1) ? Wk : Wv;
    int n = lane & 15, h = lane >> 4;
    int col = t * 16 + n;
    long fb = (long)tid * 16;
#pragma unroll
    for (int j = 0; j < 16; ++j) {
        int kk = c * 32 + h * 8 + j + (j >= 8 ? 8 : 0);
        float v = W[(long)kk * 64 + col];
        __bf16 hi = (__bf16)v;
        __bf16 lo = (__bf16)(v - (float)hi);
        whi[fb + j] = hi;
        wlo[fb + j] = lo;
    }
}

// ---------------------------------------------------------------------------
// Kernel 1: fused QKV projection.  One wave = 16 rows of x, all 3 matrices.
// hi/lo split: acc += xhi*Whi + xhi*Wlo + xlo*Whi  (~fp32 fidelity).
// ---------------------------------------------------------------------------
__global__ void __launch_bounds__(256) qkv_proj(
    const float* __restrict__ x, const __bf16* __restrict__ whi,
    const __bf16* __restrict__ wlo, const float* __restrict__ bq,
    const float* __restrict__ bk, const float* __restrict__ bv,
    float* __restrict__ qout, float* __restrict__ kout,
    float* __restrict__ vout) {
    const int lane  = threadIdx.x & 31;
    const int wid   = (blockIdx.x * blockDim.x + threadIdx.x) >> 5; // 0..2047
    const int m0    = wid * 16;
    const int n     = lane & 15;
    const int h     = lane >> 4;
    const int abase = h * 8;   // A-frag K base for this half-wave
    v8f acc[12];
#pragma unroll
    for (int i = 0; i < 12; ++i) acc[i] = v8f{};
    const float* xrow = x + (long)(m0 + n) * 1024;

    for (int c = 0; c < 32; ++c) {
        f32x4 a0 = *(const f32x4*)(xrow + c * 32 + abase);
        f32x4 a1 = *(const f32x4*)(xrow + c * 32 + abase + 4);
        f32x4 a2 = *(const f32x4*)(xrow + c * 32 + 16 + abase);
        f32x4 a3 = *(const f32x4*)(xrow + c * 32 + 16 + abase + 4);
        float av[16];
#pragma unroll
        for (int i = 0; i < 4; ++i) {
            av[i] = a0[i]; av[4 + i] = a1[i]; av[8 + i] = a2[i]; av[12 + i] = a3[i];
        }
        v16bf ah, al;
#pragma unroll
        for (int j = 0; j < 16; ++j) {
            __bf16 hi = (__bf16)av[j];
            ah[j] = hi;
            al[j] = (__bf16)(av[j] - (float)hi);
        }
#pragma unroll
        for (int mt = 0; mt < 3; ++mt) {
#pragma unroll
            for (int t = 0; t < 4; ++t) {
                long fb = (long)(((mt * 4 + t) * 32 + c) * 32 + lane) * 16;
                v16bf bh = *(const v16bf*)(whi + fb);
                v16bf bl = *(const v16bf*)(wlo + fb);
                int i = mt * 4 + t;
                acc[i] = wmma_bf16(ah, bh, acc[i]);
                acc[i] = wmma_bf16(ah, bl, acc[i]);
                acc[i] = wmma_bf16(al, bh, acc[i]);
            }
        }
    }
#pragma unroll
    for (int mt = 0; mt < 3; ++mt) {
        const float* bias = (mt == 0) ? bq : (mt == 1) ? bk : bv;
        float* outp       = (mt == 0) ? qout : (mt == 1) ? kout : vout;
#pragma unroll
        for (int t = 0; t < 4; ++t) {
            int col = t * 16 + n;
            float bb = bias[col];
            v8f a = acc[mt * 4 + t];
#pragma unroll
            for (int r = 0; r < 8; ++r) {
                long row = m0 + h * 8 + r;
                outp[row * 64 + col] = a[r] + bb;
            }
        }
    }
}

// ---------------------------------------------------------------------------
// Kernel 2: repack k and v (f32) into bf16 WMMA B-fragment blocks.
// V block per 32 keys: [t][lane][16]   (B = V[32 x 64], n-tiles t)
// K block per 32 keys: [s][c][lane][16] (B = K^T per 16-key subtile s, kchunk c)
// Each block is 2048 bf16 = 4 KB.
// ---------------------------------------------------------------------------
__global__ void __launch_bounds__(256) swizzle_kv(
    const float* __restrict__ k, const float* __restrict__ v,
    __bf16* __restrict__ kf, __bf16* __restrict__ vf) {
    int tid = blockIdx.x * blockDim.x + threadIdx.x;   // 262144 total
    int half = tid >> 17;
    int idx  = tid & 131071;
    int lane = idx & 31;
    int n = lane & 15, h = lane >> 4;
    long fb = (long)idx * 16;
    if (half == 0) {
        int t = (idx >> 5) & 3;
        int c = (idx >> 7) & 127;
        int b = idx >> 14;
        const float* vb = v + (long)b * 262144;
#pragma unroll
        for (int j = 0; j < 16; ++j) {
            int key = c * 32 + h * 8 + j + (j >= 8 ? 8 : 0);
            vf[fb + j] = (__bf16)vb[(long)key * 64 + t * 16 + n];
        }
    } else {
        int cc = (idx >> 5) & 1;
        int s  = (idx >> 6) & 1;
        int kt = (idx >> 7) & 127;
        int b  = idx >> 14;
        const float* krow = k + (long)b * 262144 +
                            (long)(kt * 32 + s * 16 + n) * 64 + cc * 32 + h * 8;
        f32x4 t0 = *(const f32x4*)(krow);
        f32x4 t1 = *(const f32x4*)(krow + 4);
        f32x4 t2 = *(const f32x4*)(krow + 16);
        f32x4 t3 = *(const f32x4*)(krow + 20);
        float tmp[16];
#pragma unroll
        for (int i = 0; i < 4; ++i) {
            tmp[i] = t0[i]; tmp[4 + i] = t1[i]; tmp[8 + i] = t2[i]; tmp[12 + i] = t3[i];
        }
#pragma unroll
        for (int j = 0; j < 16; ++j) kf[fb + j] = (__bf16)tmp[j];
    }
}

// ---------------------------------------------------------------------------
// Kernel 3: causal flash attention. Block = 128 q rows (8 waves x 16 rows).
// Per 32-key step, the shared 4KB K-frag + 4KB V-frag blocks are staged into
// LDS with double-buffered async global->LDS copies (ASYNCcnt pipeline).
// ---------------------------------------------------------------------------
__global__ void __launch_bounds__(256) attn(
    const float* __restrict__ q, const __bf16* __restrict__ kfr,
    const __bf16* __restrict__ vfr, float* __restrict__ out) {
    __shared__ __attribute__((aligned(32))) __bf16 kbufs[2][2048];
    __shared__ __attribute__((aligned(32))) __bf16 vbufs[2][2048];
    __shared__ __attribute__((aligned(32))) __bf16 psh[8][512];

    const int tid   = threadIdx.x;
    const int lane  = tid & 31;
    const int wib   = tid >> 5;
    const long rowbase = (long)blockIdx.x * 128;      // 32 blocks / batch
    const int b     = (int)(rowbase >> 12);
    const int Q0    = (int)(rowbase & 4095);
    const int q0w   = Q0 + wib * 16;                  // this wave's first q row
    const int n     = lane & 15;
    const int h     = lane >> 4;
    const int abase = h * 8;

    const __bf16* kfb = kfr + (long)b * 128 * 2048;
    const __bf16* vfb = vfr + (long)b * 128 * 2048;
    const int niter = (Q0 + 128) >> 5;

    // q A-fragments (K = H = 64 -> 2 chunks), scale folded: 0.125 * log2(e)
    v16bf qf[2];
    const float qscale = 0.125f * L2E;
    const float* qrow = q + ((long)b * 4096 + q0w + n) * 64;
#pragma unroll
    for (int c = 0; c < 2; ++c) {
        f32x4 t0 = *(const f32x4*)(qrow + c * 32 + abase);
        f32x4 t1 = *(const f32x4*)(qrow + c * 32 + abase + 4);
        f32x4 t2 = *(const f32x4*)(qrow + c * 32 + 16 + abase);
        f32x4 t3 = *(const f32x4*)(qrow + c * 32 + 16 + abase + 4);
        float tmp[16];
#pragma unroll
        for (int i = 0; i < 4; ++i) {
            tmp[i] = t0[i]; tmp[4 + i] = t1[i]; tmp[8 + i] = t2[i]; tmp[12 + i] = t3[i];
        }
#pragma unroll
        for (int j = 0; j < 16; ++j) qf[c][j] = (__bf16)(tmp[j] * qscale);
    }

    v8f acc[4];
#pragma unroll
    for (int t = 0; t < 4; ++t) acc[t] = v8f{};
    float mrow[8], lrow[8];
#pragma unroll
    for (int r = 0; r < 8; ++r) { mrow[r] = -1e30f; lrow[r] = 0.0f; }
    __bf16* pbuf = psh[wib];

    // stage K/V fragment blocks for key-tile kt into LDS buffer p
    auto stage = [&](int p, int kt) {
        const __bf16* gk = kfb + (long)(kt >> 5) * 2048 + tid * 8;
        const __bf16* gv = vfb + (long)(kt >> 5) * 2048 + tid * 8;
        stage16(gk, &kbufs[p][tid * 8]);
        stage16(gv, &vbufs[p][tid * 8]);
    };

    stage(0, 0);
    for (int it = 0; it < niter; ++it) {
        const int kt = it << 5;
        const bool hasNext = (it + 1) < niter;
        if (hasNext) stage((it + 1) & 1, (it + 1) << 5);
#if ATTN_ASYNC
        if (hasNext) __builtin_amdgcn_s_wait_asynccnt(2);
        else         __builtin_amdgcn_s_wait_asynccnt(0);
#endif
        __syncthreads();   // current buffer fully staged by all waves
        const __bf16* kcur = kbufs[it & 1];
        const __bf16* vcur = vbufs[it & 1];

        // K B-fragments straight out of LDS (pre-converted bf16)
        v16bf kf[2][2];
#pragma unroll
        for (int s = 0; s < 2; ++s)
#pragma unroll
            for (int c = 0; c < 2; ++c)
                kf[s][c] = *(const v16bf*)(kcur + ((s * 2 + c) * 32 + lane) * 16);

        // scores (log2 domain)
        v8f st[2];
#pragma unroll
        for (int s = 0; s < 2; ++s) {
            v8f z = v8f{};
            z = wmma_bf16(qf[0], kf[s][0], z);
            z = wmma_bf16(qf[1], kf[s][1], z);
            st[s] = z;
        }
        // causal mask
#pragma unroll
        for (int s = 0; s < 2; ++s) {
            int col = kt + s * 16 + n;
#pragma unroll
            for (int r = 0; r < 8; ++r) {
                int row = q0w + h * 8 + r;
                if (col > row) st[s][r] = -1e30f;
            }
        }
        // online softmax
        float al[8];
#pragma unroll
        for (int r = 0; r < 8; ++r) {
            float mx = fmaxf(st[0][r], st[1][r]);
            mx = fmaxf(mx, __shfl_xor(mx, 1, 32));
            mx = fmaxf(mx, __shfl_xor(mx, 2, 32));
            mx = fmaxf(mx, __shfl_xor(mx, 4, 32));
            mx = fmaxf(mx, __shfl_xor(mx, 8, 32));
            float mn = fmaxf(mrow[r], mx);
            al[r] = __builtin_exp2f(mrow[r] - mn);
            mrow[r] = mn;
        }
#pragma unroll
        for (int r = 0; r < 8; ++r) {
            float p0 = __builtin_exp2f(st[0][r] - mrow[r]);
            float p1 = __builtin_exp2f(st[1][r] - mrow[r]);
            st[0][r] = p0; st[1][r] = p1;
            float s2 = p0 + p1;
            s2 += __shfl_xor(s2, 1, 32);
            s2 += __shfl_xor(s2, 2, 32);
            s2 += __shfl_xor(s2, 4, 32);
            s2 += __shfl_xor(s2, 8, 32);
            lrow[r] = lrow[r] * al[r] + s2;
        }
#pragma unroll
        for (int t = 0; t < 4; ++t)
#pragma unroll
            for (int r = 0; r < 8; ++r) acc[t][r] *= al[r];

        // P: C-layout -> A-layout via per-wave LDS tile (DS ops in-order)
#pragma unroll
        for (int s = 0; s < 2; ++s)
#pragma unroll
            for (int r = 0; r < 8; ++r)
                pbuf[(h * 8 + r) * 32 + s * 16 + n] = (__bf16)st[s][r];
        v16bf pf;
        {
            v8bf lo8 = *(const v8bf*)(pbuf + n * 32 + abase);
            v8bf hi8 = *(const v8bf*)(pbuf + n * 32 + 16 + abase);
#pragma unroll
            for (int j = 0; j < 8; ++j) { pf[j] = lo8[j]; pf[8 + j] = hi8[j]; }
        }
        // PV
#pragma unroll
        for (int t = 0; t < 4; ++t) {
            v16bf vfrag = *(const v16bf*)(vcur + (t * 32 + lane) * 16);
            acc[t] = wmma_bf16(pf, vfrag, acc[t]);
        }
        __syncthreads();   // all waves done reading cur before it is re-staged
    }

    float* ob = out + (long)b * 4096 * 64;
#pragma unroll
    for (int r = 0; r < 8; ++r) {
        float inv = 1.0f / lrow[r];
        int row = q0w + h * 8 + r;
#pragma unroll
        for (int t = 0; t < 4; ++t)
            ob[(long)row * 64 + t * 16 + n] = acc[t][r] * inv;
    }
}

// ---------------------------------------------------------------------------
extern "C" void kernel_launch(void* const* d_in, const int* in_sizes, int n_in,
                              void* d_out, int out_size, void* d_ws, size_t ws_size,
                              hipStream_t stream) {
    const float* x  = (const float*)d_in[0];
    const float* Wq = (const float*)d_in[1];
    const float* bq = (const float*)d_in[2];
    const float* Wk = (const float*)d_in[3];
    const float* bk = (const float*)d_in[4];
    const float* Wv = (const float*)d_in[5];
    const float* bv = (const float*)d_in[6];
    float* out  = (float*)d_out;
    float* kout = out + 2097152;   // B*S*H
    float* vout = out + 4194304;

    char* ws = (char*)d_ws;
    __bf16* whi = (__bf16*)ws;                               // 384 KB
    __bf16* wlo = (__bf16*)(ws + 393216);                    // 384 KB
    __bf16* vfr = (__bf16*)(ws + 786432);                    // 4 MB
    __bf16* kfr = (__bf16*)(ws + 786432 + 4194304);          // 4 MB
    float*  qws = (float*)(ws + 786432 + 2 * 4194304);       // 8 MB

    prep_w    <<<48,   256, 0, stream>>>(Wq, Wk, Wv, whi, wlo);
    qkv_proj  <<<256,  256, 0, stream>>>(x, whi, wlo, bq, bk, bv, qws, kout, vout);
    swizzle_kv<<<1024, 256, 0, stream>>>(kout, vout, kfr, vfr);
    attn      <<<256,  256, 0, stream>>>(qws, kfr, vfr, out);
}
